// ReadoutFunction_34797825032694
// MI455X (gfx1250) — matrix operand order repaired and checked by reading, not compiled, not run on hardware
//
#include <hip/hip_runtime.h>

typedef __attribute__((ext_vector_type(16))) __bf16 v16bf;
typedef __attribute__((ext_vector_type(8)))  __bf16 v8bf;
typedef __attribute__((ext_vector_type(8)))  float  v8f;
typedef __attribute__((ext_vector_type(4)))  float  v4f;

#define HIDDEN  128
#define NGRAPHS 8192
#define OUTD    16

__device__ __forceinline__ __bf16 bf(float x) { return (__bf16)x; }

__device__ __forceinline__ v16bf catbf(v8bf lo, v8bf hi) {
  return __builtin_shufflevector(lo, hi, 0,1,2,3,4,5,6,7,8,9,10,11,12,13,14,15);
}

// lanes 0-15 get lane+16 (16-31 keep own) in one DS op: group-of-32 swizzle, or=0x10, and=0x1f
__device__ __forceinline__ float swap16(float x) {
  int r = __builtin_amdgcn_ds_swizzle(__builtin_bit_cast(int, x), 0x021F);
  return __builtin_bit_cast(float, r);
}

__device__ __forceinline__ v16bf cvt16(v4f f0, v4f f1, v4f f2, v4f f3) {
  v16bf a;
  a[0]=bf(f0[0]);  a[1]=bf(f0[1]);  a[2]=bf(f0[2]);  a[3]=bf(f0[3]);
  a[4]=bf(f1[0]);  a[5]=bf(f1[1]);  a[6]=bf(f1[2]);  a[7]=bf(f1[3]);
  a[8]=bf(f2[0]);  a[9]=bf(f2[1]);  a[10]=bf(f2[2]); a[11]=bf(f2[3]);
  a[12]=bf(f3[0]); a[13]=bf(f3[1]); a[14]=bf(f3[2]); a[15]=bf(f3[3]);
  return a;
}

// ---------------------------------------------------------------------------
// Prep: transpose + f32->bf16 convert weights into scratch (tiny).
//   trans_w [k=128][n=128] row-major  -> Wt [n][k] bf16
//   out_w   [k=128][n=16 ] row-major  -> Ot [n][k] bf16
// ---------------------------------------------------------------------------
__global__ __launch_bounds__(256) void prep_weights(
    const float* __restrict__ trans_w, const float* __restrict__ out_w,
    __bf16* __restrict__ Wt, __bf16* __restrict__ Ot)
{
  int tid = blockIdx.x * blockDim.x + threadIdx.x;
  int stride = gridDim.x * blockDim.x;
  for (int i = tid; i < HIDDEN * HIDDEN; i += stride) {
    int n = i >> 7, k = i & (HIDDEN - 1);
    Wt[i] = bf(trans_w[k * HIDDEN + n]);
  }
  for (int i = tid; i < OUTD * HIDDEN; i += stride) {
    int n = i >> 7, k = i & (HIDDEN - 1);
    Ot[i] = bf(out_w[k * OUTD + n]);
  }
}

// ---------------------------------------------------------------------------
// Fused gate + transform + segment-sum pooling.
// One wave per 16-node slab; bf16 WMMA 16x16x32; sorted-segment run
// compression with f32 atomics into pooled[8192][128].
// ---------------------------------------------------------------------------
__global__ __launch_bounds__(256) void fused_gate_pool(
    const float* __restrict__ h, const int* __restrict__ bidx,
    const float* __restrict__ gate_w, const float* __restrict__ gate_b,
    const __bf16* __restrict__ Wt, const float* __restrict__ trans_b,
    float* __restrict__ pooled, int n_nodes)
{
  const int lane = threadIdx.x & 31;
  const int wave = threadIdx.x >> 5;
  const long slab = (long)blockIdx.x * 8 + wave;
  const long base = slab * 16;
  if (base >= n_nodes) return;           // wave-uniform: EXEC stays all-ones

  const int half = lane >> 4;            // 0: lanes 0-15, 1: lanes 16-31
  const int l16  = lane & 15;
  const float* hrow = h + (size_t)(base + l16) * HIDDEN;

  // --- load A fragments (4 K-steps of 32, nontemporal stream) + gate dot ---
  v16bf A[4];
  float gpart = 0.f;
#pragma unroll
  for (int t = 0; t < 4; ++t) {
    const int k0 = t * 32 + half * 8;    // this lane's K groups: k0..k0+7, k0+16..k0+23
    v4f f0 = __builtin_nontemporal_load((const v4f*)(hrow + k0));
    v4f f1 = __builtin_nontemporal_load((const v4f*)(hrow + k0 + 4));
    v4f f2 = __builtin_nontemporal_load((const v4f*)(hrow + k0 + 16));
    v4f f3 = __builtin_nontemporal_load((const v4f*)(hrow + k0 + 20));
    v4f w0 = *(const v4f*)(gate_w + k0);
    v4f w1 = *(const v4f*)(gate_w + k0 + 4);
    v4f w2 = *(const v4f*)(gate_w + k0 + 16);
    v4f w3 = *(const v4f*)(gate_w + k0 + 20);
    v4f d = f0 * w0 + f1 * w1 + f2 * w2 + f3 * w3;
    gpart += d[0] + d[1] + d[2] + d[3];
    A[t] = cvt16(f0, f1, f2, f3);
  }

  // --- finish gate: lane l and l+16 hold the two halves of row l%16 --------
  float gsum = gpart + swap16(gpart);
  float gate = __builtin_amdgcn_rcpf(1.f + __expf(-(gsum + gate_b[0])));

  // per-register-row gate value + segment id (D layout: VGPR r holds
  // row r for lanes 0-15, row r+8 for lanes 16-31)
  float g[8]; int s[8];
#pragma unroll
  for (int r = 0; r < 8; ++r) {
    g[r] = __shfl(gate, r + half * 8, 32);
    s[r] = bidx[base + r + half * 8];
  }

  // --- 8 N-tiles of 16 columns --------------------------------------------
#pragma unroll
  for (int nt = 0; nt < 8; ++nt) {
    const int n = nt * 16 + l16;
    const float bias = trans_b[n];
    v8f C;
#pragma unroll
    for (int i = 0; i < 8; ++i) C[i] = bias;

    const __bf16* wr = Wt + (size_t)n * HIDDEN;
#pragma unroll
    for (int t = 0; t < 4; ++t) {
      // B layout: lanes 0-15 hold K 0-7,16-23, lanes 16-31 hold K 8-15,24-31 (2/VGPR)
      const int k0 = t * 32 + half * 8;
      v8bf blo = *(const v8bf*)(wr + k0);
      v8bf bhi = *(const v8bf*)(wr + k0 + 16);
      v16bf B = catbf(blo, bhi);
      C = __builtin_amdgcn_wmma_f32_16x16x32_bf16(
            false, A[t], false, B, (short)0, C, false, false);
    }

    // gate-scale + sorted-segment run compression, flush with f32 atomics
    float acc = C[0] * g[0];
    int cur = s[0];
#pragma unroll
    for (int r = 1; r < 8; ++r) {
      float v = C[r] * g[r];
      if (s[r] == cur) {
        acc += v;
      } else {
        atomicAdd(&pooled[(size_t)cur * HIDDEN + n], acc);
        cur = s[r]; acc = v;
      }
    }
    atomicAdd(&pooled[(size_t)cur * HIDDEN + n], acc);
  }
}

// ---------------------------------------------------------------------------
// Output GEMM: pooled[8192][128] @ out_w[128][16] + out_b -> out[8192][16].
// One wave per 16-graph tile; 4 bf16 WMMAs (K=128).
// ---------------------------------------------------------------------------
__global__ __launch_bounds__(256) void out_gemm(
    const float* __restrict__ pooled, const __bf16* __restrict__ Ot,
    const float* __restrict__ out_b, float* __restrict__ out)
{
  const int lane = threadIdx.x & 31;
  const int wave = threadIdx.x >> 5;
  const int tile = blockIdx.x * 8 + wave;      // 512 tiles total
  const int rbase = tile * 16;
  const int half = lane >> 4;
  const int l16  = lane & 15;
  const float* prow = pooled + (size_t)(rbase + l16) * HIDDEN;

  v16bf A[4];
#pragma unroll
  for (int t = 0; t < 4; ++t) {
    const int k0 = t * 32 + half * 8;
    v4f f0 = *(const v4f*)(prow + k0);
    v4f f1 = *(const v4f*)(prow + k0 + 4);
    v4f f2 = *(const v4f*)(prow + k0 + 16);
    v4f f3 = *(const v4f*)(prow + k0 + 20);
    A[t] = cvt16(f0, f1, f2, f3);
  }

  const float bias = out_b[l16];
  v8f C;
#pragma unroll
  for (int i = 0; i < 8; ++i) C[i] = bias;

  const __bf16* wr = Ot + (size_t)l16 * HIDDEN;
#pragma unroll
  for (int t = 0; t < 4; ++t) {
    const int k0 = t * 32 + half * 8;
    v8bf blo = *(const v8bf*)(wr + k0);
    v8bf bhi = *(const v8bf*)(wr + k0 + 16);
    v16bf B = catbf(blo, bhi);
    C = __builtin_amdgcn_wmma_f32_16x16x32_bf16(
          false, A[t], false, B, (short)0, C, false, false);
  }

#pragma unroll
  for (int r = 0; r < 8; ++r)
    __builtin_nontemporal_store(C[r],
        out + (size_t)(rbase + r + half * 8) * OUTD + l16);
}

// ---------------------------------------------------------------------------
extern "C" void kernel_launch(void* const* d_in, const int* in_sizes, int n_in,
                              void* d_out, int out_size, void* d_ws, size_t ws_size,
                              hipStream_t stream) {
  const float* h       = (const float*)d_in[0];
  const int*   bidx    = (const int*)  d_in[1];
  const float* gate_w  = (const float*)d_in[2];
  const float* gate_b  = (const float*)d_in[3];
  const float* trans_w = (const float*)d_in[4];
  const float* trans_b = (const float*)d_in[5];
  const float* out_w   = (const float*)d_in[6];
  const float* out_b   = (const float*)d_in[7];
  float* out = (float*)d_out;
  const int n_nodes = in_sizes[0] / HIDDEN;

  // scratch layout: pooled f32 [8192*128] | Wt bf16 [128*128] | Ot bf16 [16*128]
  float* pooled = (float*)d_ws;
  const size_t pooled_bytes = (size_t)NGRAPHS * HIDDEN * sizeof(float);
  __bf16* Wt = (__bf16*)((char*)d_ws + pooled_bytes);
  __bf16* Ot = Wt + HIDDEN * HIDDEN;

  (void)hipMemsetAsync(d_ws, 0, pooled_bytes, stream);
  prep_weights<<<16, 256, 0, stream>>>(trans_w, out_w, Wt, Ot);

  const int slabs  = (n_nodes + 15) / 16;
  const int blocks = (slabs + 7) / 8;
  fused_gate_pool<<<blocks, 256, 0, stream>>>(h, bidx, gate_w, gate_b, Wt,
                                              trans_b, pooled, n_nodes);
  out_gemm<<<NGRAPHS / 16 / 8, 256, 0, stream>>>(pooled, Ot, out_b, out);
}
